// SequencePredictionModel_71683004170518
// MI455X (gfx1250) — matrix operand order repaired and checked by reading, not compiled, not run on hardware
//
#include <hip/hip_runtime.h>
#include <hip/hip_bf16.h>

// ---------------------------------------------------------------------------
// GNN stack on MI455X (gfx1250, wave32, WMMA + Tensor Data Mover).
//  - aggregation is linear: mean_agg(x) @ W == mean_agg(x @ W) -> scatter in
//    256-dim space (16.8M atomics instead of 268M).
//  - one-time fp32->bf16 conversion of x / weights (weights pre-transposed to
//    [N,K]) so GEMM staging is a contiguous copy: TDM tensor_load_to_lds with
//    double buffering, v_wmma_f32_16x16x32_bf16 compute, f32 accumulate.
// ---------------------------------------------------------------------------

typedef __attribute__((ext_vector_type(16))) __bf16        v16bf;
typedef __attribute__((ext_vector_type(8)))  float         v8f;
typedef __attribute__((ext_vector_type(4)))  float         f32x4;
typedef __attribute__((ext_vector_type(4)))  unsigned int  u32x4;
typedef __attribute__((ext_vector_type(2)))  unsigned int  u32x2;
typedef __attribute__((ext_vector_type(8)))  int           i32x8;
typedef __attribute__((ext_vector_type(4)))  int           i32x4;

#define BM 128
#define BN 64
#define BK 64   // two WMMA K-steps per staged chunk

#if defined(__has_builtin)
#if __has_builtin(__builtin_amdgcn_tensor_load_to_lds) && __has_builtin(__builtin_amdgcn_s_wait_tensorcnt)
#define USE_TDM 1
#endif
#endif

union FragBF { v16bf v; u32x4 q[2]; };

static __device__ __forceinline__ unsigned short f2bf(float f) {
    unsigned int u = __builtin_bit_cast(unsigned int, f);
    u += 0x7FFFu + ((u >> 16) & 1u);            // RNE
    return (unsigned short)(u >> 16);
}

#ifdef USE_TDM
// Issue one TDM 2-D tile load: tile_d1 rows x tile_d0 bf16 elements,
// global row stride = stride0 elements, packed contiguously into LDS.
static __device__ __forceinline__ void tdm_load_tile(const unsigned short* gptr,
                                                     unsigned int lds_byte,
                                                     int stride0, int tile_d0, int tile_d1)
{
    unsigned long long ga = (unsigned long long)(uintptr_t)gptr;
    u32x4 g0;
    g0[0] = 1u;                                            // count=1 (valid D#)
    g0[1] = lds_byte;                                      // lds_addr
    g0[2] = (unsigned int)(ga & 0xFFFFFFFFu);              // global_addr[31:0]
    g0[3] = (unsigned int)((ga >> 32) & 0x01FFFFFFu)       // global_addr[56:32]
          | (2u << 30);                                    // type=2 ("image")
    i32x8 g1;
    g1[0] = (int)(1u << 16);                               // wg_mask=0, data_size=1 (2B)
    g1[1] = (int)(((unsigned)tile_d0 & 0xFFFFu) << 16);    // tensor_dim0 = tile_d0
    g1[2] = (int)((((unsigned)tile_d0 >> 16) & 0xFFFFu)
          | (((unsigned)tile_d1 & 0xFFFFu) << 16));        // tensor_dim1 = tile_d1
    g1[3] = (int)((((unsigned)tile_d1 >> 16) & 0xFFFFu)
          | (((unsigned)tile_d0 & 0xFFFFu) << 16));        // tile_dim0
    g1[4] = (int)((unsigned)tile_d1 & 0xFFFFu);            // tile_dim1, tile_dim2=0
    g1[5] = stride0;                                       // tensor_dim0_stride[31:0]
    g1[6] = 0;                                             // stride0[47:32]=0, stride1=0
    g1[7] = 0;
    i32x4 z4 = {0, 0, 0, 0};
#if __clang_major__ >= 23
    i32x8 z8 = {0, 0, 0, 0, 0, 0, 0, 0};
    __builtin_amdgcn_tensor_load_to_lds(g0, g1, z4, z4, z8, 0);
#else
    __builtin_amdgcn_tensor_load_to_lds(g0, g1, z4, z4, 0);
#endif
}
#endif

// C[M,N] = act(A[M,K] @ Bt[N,K]^T + bias).  A,Bt bf16.  M%128==0, N%64==0, K%64==0.
__global__ __launch_bounds__(256)
void gemm_bf16_wmma(const unsigned short* __restrict__ A,
                    const unsigned short* __restrict__ Bt,
                    const float* __restrict__ bias,
                    float* __restrict__ Cf, unsigned short* __restrict__ Cbf,
                    int M, int N, int K, int relu)
{
    __shared__ __align__(16) unsigned short lA[2][BM * BK];   // [row][k]
    __shared__ __align__(16) unsigned short lB[2][BN * BK];   // [col][k]

    const int t     = threadIdx.x;
    const int lane  = t & 31;
    const int wave  = t >> 5;
    const int wm    = (wave & 3) << 5;
    const int wn    = (wave >> 2) << 5;
    const int m0    = blockIdx.y * BM;
    const int n0    = blockIdx.x * BN;
    const int lrow  = lane & 15;
    const int khalf = lane >> 4;
    const int nk    = K / BK;

    v8f acc[2][2];
#pragma unroll
    for (int i = 0; i < 2; ++i)
#pragma unroll
        for (int j = 0; j < 2; ++j)
            acc[i][j] = v8f{0.f, 0.f, 0.f, 0.f, 0.f, 0.f, 0.f, 0.f};

#ifdef USE_TDM
    if (wave == 0) {
        tdm_load_tile(A + (size_t)m0 * K, (unsigned int)(uintptr_t)(void*)&lA[0][0], K, BK, BM);
        tdm_load_tile(Bt + (size_t)n0 * K, (unsigned int)(uintptr_t)(void*)&lB[0][0], K, BK, BN);
    }
    for (int i = 0; i < nk; ++i) {
        const int buf = i & 1;
        if (wave == 0) {
            if (i + 1 < nk) {
                const int k1 = (i + 1) * BK;
                tdm_load_tile(A + (size_t)m0 * K + k1,
                              (unsigned int)(uintptr_t)(void*)&lA[buf ^ 1][0], K, BK, BM);
                tdm_load_tile(Bt + (size_t)n0 * K + k1,
                              (unsigned int)(uintptr_t)(void*)&lB[buf ^ 1][0], K, BK, BN);
                __builtin_amdgcn_s_wait_tensorcnt(2);   // tiles for chunk i complete
            } else {
                __builtin_amdgcn_s_wait_tensorcnt(0);
            }
        }
        __syncthreads();
#pragma unroll
        for (int ks = 0; ks < 2; ++ks) {
            FragBF fa[2], fb[2];
#pragma unroll
            for (int mt = 0; mt < 2; ++mt) {
                const unsigned short* pa = &lA[buf][(wm + mt * 16 + lrow) * BK + ks * 32];
                fa[mt].q[0] = *(const u32x4*)(pa + 8 * khalf);
                fa[mt].q[1] = *(const u32x4*)(pa + 16 + 8 * khalf);
            }
#pragma unroll
            for (int nt = 0; nt < 2; ++nt) {
                const unsigned short* pb = &lB[buf][(wn + nt * 16 + lrow) * BK + ks * 32];
                fb[nt].q[0] = *(const u32x4*)(pb + 8 * khalf);
                fb[nt].q[1] = *(const u32x4*)(pb + 16 + 8 * khalf);
            }
#pragma unroll
            for (int mt = 0; mt < 2; ++mt)
#pragma unroll
                for (int nt = 0; nt < 2; ++nt)
                    acc[mt][nt] = __builtin_amdgcn_wmma_f32_16x16x32_bf16(
                        false, fa[mt].v, false, fb[nt].v,
                        (short)0, acc[mt][nt], false, false);
        }
        __syncthreads();
    }
#else
    // Fallback: manual staging (pure bf16 copies, no conversion in loop)
    const int arow = t >> 1, ahalf = (t & 1) * 32;    // A: half-row (64B) per thread
    const int brow = t >> 2, bq = (t & 3) * 16;       // B: quarter-row (32B) per thread
    for (int i = 0; i < nk; ++i) {
        const int k0 = i * BK;
        {
            const u32x4* gp = (const u32x4*)(A + (size_t)(m0 + arow) * K + k0 + ahalf);
            u32x4* lp = (u32x4*)&lA[0][arow * BK + ahalf];
#pragma unroll
            for (int j = 0; j < 4; ++j) lp[j] = gp[j];
            const u32x4* gq = (const u32x4*)(Bt + (size_t)(n0 + brow) * K + k0 + bq);
            u32x4* lq = (u32x4*)&lB[0][brow * BK + bq];
            lq[0] = gq[0]; lq[1] = gq[1];
        }
        __syncthreads();
#pragma unroll
        for (int ks = 0; ks < 2; ++ks) {
            FragBF fa[2], fb[2];
#pragma unroll
            for (int mt = 0; mt < 2; ++mt) {
                const unsigned short* pa = &lA[0][(wm + mt * 16 + lrow) * BK + ks * 32];
                fa[mt].q[0] = *(const u32x4*)(pa + 8 * khalf);
                fa[mt].q[1] = *(const u32x4*)(pa + 16 + 8 * khalf);
            }
#pragma unroll
            for (int nt = 0; nt < 2; ++nt) {
                const unsigned short* pb = &lB[0][(wn + nt * 16 + lrow) * BK + ks * 32];
                fb[nt].q[0] = *(const u32x4*)(pb + 8 * khalf);
                fb[nt].q[1] = *(const u32x4*)(pb + 16 + 8 * khalf);
            }
#pragma unroll
            for (int mt = 0; mt < 2; ++mt)
#pragma unroll
                for (int nt = 0; nt < 2; ++nt)
                    acc[mt][nt] = __builtin_amdgcn_wmma_f32_16x16x32_bf16(
                        false, fa[mt].v, false, fb[nt].v,
                        (short)0, acc[mt][nt], false, false);
        }
        __syncthreads();
    }
#endif

    // epilogue: C/D layout -> col = lane&15, row = r + 8*(lane>>4)
#pragma unroll
    for (int mt = 0; mt < 2; ++mt)
#pragma unroll
        for (int nt = 0; nt < 2; ++nt) {
            const int col = n0 + wn + nt * 16 + lrow;
            const float bb = bias ? bias[col] : 0.0f;
#pragma unroll
            for (int r = 0; r < 8; ++r) {
                const int row = m0 + wm + mt * 16 + r + 8 * khalf;
                float v = acc[mt][nt][r] + bb;
                if (relu) v = fmaxf(v, 0.0f);
                if (Cbf) Cbf[(size_t)row * N + col] = f2bf(v);
                else     Cf[(size_t)row * N + col]  = v;
            }
        }
}

// ---------------- conversion kernels (one-time) ----------------

__global__ void cvt_bf16_vec(const float* __restrict__ in, unsigned short* __restrict__ out, int n4) {
    int i = blockIdx.x * blockDim.x + threadIdx.x;
    if (i < n4) {
        f32x4 f = ((const f32x4*)in)[i];
        u32x2 o;
        o.x = ((unsigned)f2bf(f.y) << 16) | f2bf(f.x);
        o.y = ((unsigned)f2bf(f.w) << 16) | f2bf(f.z);
        ((u32x2*)out)[i] = o;
    }
}

// W[K,N] fp32 -> Wt[N,K] bf16
__global__ void cvt_transpose(const float* __restrict__ W, unsigned short* __restrict__ Wt,
                              int K, int N) {
    int i = blockIdx.x * blockDim.x + threadIdx.x;
    if (i < K * N) {
        int k = i / N, n = i - k * N;
        Wt[(size_t)n * K + k] = f2bf(W[i]);
    }
}

// ---------------- graph kernels ----------------

__global__ void deg_kernel(const int* __restrict__ dst, float* __restrict__ deg, int E) {
    int e = blockIdx.x * blockDim.x + threadIdx.x;
    if (e < E) atomicAdd(&deg[dst[e]], 1.0f);
}

__global__ void dinv_kernel(const float* __restrict__ deg, float* __restrict__ dinv, int n) {
    int v = blockIdx.x * blockDim.x + threadIdx.x;
    if (v < n) dinv[v] = rsqrtf(deg[v] + 1.0f);       // GCN degree includes self-loop
}

__global__ void scatter_sum(const int* __restrict__ src, const int* __restrict__ dst,
                            const float* __restrict__ feat, float* __restrict__ out, int dim) {
    const int e = blockIdx.x, f = threadIdx.x;
    atomicAdd(&out[(size_t)dst[e] * dim + f], feat[(size_t)src[e] * dim + f]);
}

__global__ void scatter_norm(const int* __restrict__ src, const int* __restrict__ dst,
                             const float* __restrict__ dinv, const float* __restrict__ feat,
                             float* __restrict__ out, int dim) {
    const int e = blockIdx.x, f = threadIdx.x;
    const int s = src[e], d = dst[e];
    atomicAdd(&out[(size_t)d * dim + f], dinv[s] * dinv[d] * feat[(size_t)s * dim + f]);
}

// h_bf16 = relu(agg/max(deg,1) + bl + xwr)
__global__ void sage_combine(const float* __restrict__ agg, const float* __restrict__ deg,
                             const float* __restrict__ bl, const float* __restrict__ xwr,
                             unsigned short* __restrict__ h, int dim) {
    const int v = blockIdx.x, f = threadIdx.x;
    const float dd = fmaxf(deg[v], 1.0f);
    const size_t i = (size_t)v * dim + f;
    h[i] = f2bf(fmaxf(agg[i] / dd + bl[f] + xwr[i], 0.0f));
}

// h_bf16 = relu(acc + dinv[v]^2 * hw[v] + b)   (self-loop folded in)
__global__ void gcn_combine(const float* __restrict__ acc, const float* __restrict__ dinv,
                            const float* __restrict__ hw, const float* __restrict__ b,
                            unsigned short* __restrict__ h, int dim) {
    const int v = blockIdx.x, f = threadIdx.x;
    const float di = dinv[v];
    const size_t i = (size_t)v * dim + f;
    h[i] = f2bf(fmaxf(acc[i] + di * di * hw[i] + b[f], 0.0f));
}

// ---------------- launch ----------------

extern "C" void kernel_launch(void* const* d_in, const int* in_sizes, int n_in,
                              void* d_out, int out_size, void* d_ws, size_t ws_size,
                              hipStream_t stream) {
    const int F = 4096, E = 65536, D1 = 256, D2 = 128, NOUT = 4 * F;

    const float* x       = (const float*)d_in[0];
    const int*   ei      = (const int*)d_in[1];
    const int*   src     = ei;
    const int*   dst     = ei + E;
    const float* sage_Wl = (const float*)d_in[2];
    const float* sage_bl = (const float*)d_in[3];
    const float* sage_Wr = (const float*)d_in[4];
    const float* gcn1_W  = (const float*)d_in[5];
    const float* gcn1_b  = (const float*)d_in[6];
    const float* gcn2_W  = (const float*)d_in[7];
    const float* gcn2_b  = (const float*)d_in[8];
    const float* fc1_W   = (const float*)d_in[9];
    const float* fc1_b   = (const float*)d_in[10];
    const float* fc2_W   = (const float*)d_in[11];
    const float* fc2_b   = (const float*)d_in[12];
    const float* out_W   = (const float*)d_in[13];
    const float* out_b   = (const float*)d_in[14];

    char* cur = (char*)d_ws;
    auto alloc = [&](size_t bytes) { char* r = cur; cur += (bytes + 255) & ~(size_t)255; return r; };

    float*          deg   = (float*)alloc(F * 4);
    float*          dinv  = (float*)alloc(F * 4);
    unsigned short* x_bf  = (unsigned short*)alloc((size_t)F * F * 2);
    unsigned short* Wl_t  = (unsigned short*)alloc((size_t)F * D1 * 2);
    unsigned short* Wr_t  = (unsigned short*)alloc((size_t)F * D1 * 2);
    unsigned short* g1_t  = (unsigned short*)alloc((size_t)D1 * D1 * 2);
    unsigned short* g2_t  = (unsigned short*)alloc((size_t)D1 * D2 * 2);
    unsigned short* f1_t  = (unsigned short*)alloc((size_t)D2 * D2 * 2);
    unsigned short* f2_t  = (unsigned short*)alloc((size_t)D2 * D2 * 2);
    unsigned short* oW_t  = (unsigned short*)alloc((size_t)D2 * NOUT * 2);
    float*          xw_l  = (float*)alloc((size_t)F * D1 * 4);
    float*          xw_r  = (float*)alloc((size_t)F * D1 * 4);
    float*          agg   = (float*)alloc((size_t)F * D1 * 4);
    float*          hw    = (float*)alloc((size_t)F * D1 * 4);
    unsigned short* hbf0  = (unsigned short*)alloc((size_t)F * D1 * 2);
    unsigned short* hbf1  = (unsigned short*)alloc((size_t)F * D1 * 2);

    dim3 blk(256);

    // one-time precision/layout conversion
    cvt_bf16_vec<<<(F * F / 4 + 255) / 256, blk, 0, stream>>>(x, x_bf, F * F / 4);
    cvt_transpose<<<(F * D1 + 255) / 256, blk, 0, stream>>>(sage_Wl, Wl_t, F, D1);
    cvt_transpose<<<(F * D1 + 255) / 256, blk, 0, stream>>>(sage_Wr, Wr_t, F, D1);
    cvt_transpose<<<(D1 * D1 + 255) / 256, blk, 0, stream>>>(gcn1_W, g1_t, D1, D1);
    cvt_transpose<<<(D1 * D2 + 255) / 256, blk, 0, stream>>>(gcn2_W, g2_t, D1, D2);
    cvt_transpose<<<(D2 * D2 + 255) / 256, blk, 0, stream>>>(fc1_W, f1_t, D2, D2);
    cvt_transpose<<<(D2 * D2 + 255) / 256, blk, 0, stream>>>(fc2_W, f2_t, D2, D2);
    cvt_transpose<<<(D2 * NOUT + 255) / 256, blk, 0, stream>>>(out_W, oW_t, D2, NOUT);

    hipMemsetAsync(deg, 0, F * sizeof(float), stream);
    hipMemsetAsync(agg, 0, (size_t)F * D1 * sizeof(float), stream);
    deg_kernel<<<E / 256, blk, 0, stream>>>(dst, deg, E);
    dinv_kernel<<<F / 256, blk, 0, stream>>>(deg, dinv, F);

    // SAGE (aggregate after projection)
    gemm_bf16_wmma<<<dim3(D1 / BN, F / BM), blk, 0, stream>>>(x_bf, Wl_t, nullptr, xw_l, nullptr, F, D1, F, 0);
    gemm_bf16_wmma<<<dim3(D1 / BN, F / BM), blk, 0, stream>>>(x_bf, Wr_t, nullptr, xw_r, nullptr, F, D1, F, 0);
    scatter_sum<<<E, D1, 0, stream>>>(src, dst, xw_l, agg, D1);
    sage_combine<<<F, D1, 0, stream>>>(agg, deg, sage_bl, xw_r, hbf0, D1);

    // GCN1 (256 -> 256)
    gemm_bf16_wmma<<<dim3(D1 / BN, F / BM), blk, 0, stream>>>(hbf0, g1_t, nullptr, hw, nullptr, F, D1, D1, 0);
    hipMemsetAsync(agg, 0, (size_t)F * D1 * sizeof(float), stream);
    scatter_norm<<<E, D1, 0, stream>>>(src, dst, dinv, hw, agg, D1);
    gcn_combine<<<F, D1, 0, stream>>>(agg, dinv, hw, gcn1_b, hbf0, D1);

    // GCN2 (256 -> 128)
    gemm_bf16_wmma<<<dim3(D2 / BN, F / BM), blk, 0, stream>>>(hbf0, g2_t, nullptr, hw, nullptr, F, D2, D1, 0);
    hipMemsetAsync(agg, 0, (size_t)F * D2 * sizeof(float), stream);
    scatter_norm<<<E, D2, 0, stream>>>(src, dst, dinv, hw, agg, D2);
    gcn_combine<<<F, D2, 0, stream>>>(agg, dinv, hw, gcn2_b, hbf0, D2);

    // MLP head (bf16 activations end to end)
    gemm_bf16_wmma<<<dim3(D2 / BN, F / BM), blk, 0, stream>>>(hbf0, f1_t, fc1_b, nullptr, hbf1, F, D2, D2, 1);
    gemm_bf16_wmma<<<dim3(D2 / BN, F / BM), blk, 0, stream>>>(hbf1, f2_t, fc2_b, nullptr, hbf0, F, D2, D2, 1);

    // logits: [4096x128] @ [128x16384] + bias -> 256 MB fp32 (traffic floor)
    gemm_bf16_wmma<<<dim3(NOUT / BN, F / BM), blk, 0, stream>>>(hbf0, oW_t, out_b, (float*)d_out, nullptr, F, NOUT, D2, 0);
}
